// SoftAlign_74071005987587
// MI455X (gfx1250) — matrix operand order repaired and checked by reading, not compiled, not run on hardware
//
#include <hip/hip_runtime.h>
#include <hip/hip_bf16.h>

// ---------------------------------------------------------------------------
// SoftAlign (cosine cross-attention) for MI455X / gfx1250, wave32 + WMMA bf16.
//   q = l2norm(Q); p = l2norm(P); O = softmax(q p^T / 0.07) P
// Fused flash-attention: S never hits HBM. bf16 WMMA, f32 accumulate.
// Tile staging now uses CDNA5 async global->LDS DMA (GLOBAL_LOAD_ASYNC_TO_LDS,
// ASYNCcnt) with double-buffered LDS so tile j+1 streams in during the 64
// WMMAs of tile j.
// Workspace layout (d_ws, ~50.3 MB):
//   qn  [B][NQ][D]  bf16  (normalized query)
//   pn  [B][NP][D]  bf16  (normalized prompt)
//   vt  [B][D][NP]  bf16  (raw prompt, transposed -> contiguous-K B-frags)
// ---------------------------------------------------------------------------

typedef __attribute__((ext_vector_type(16))) __bf16 v16bf;
typedef __attribute__((ext_vector_type(8)))  float  v8f;

#define BDIM 8
#define NQ   2048
#define NP   2048
#define DIM  512
#define TAU  0.07f

#define QT     64      // query rows per workgroup (4 waves x 16)
#define NWAVE  4
#define KT     32      // prompt rows per iteration (= one WMMA K step for PV)
#define PSTR   520     // LDS P-tile row stride in bf16 elems (512 + 8 pad)
#define VSTR   40      // LDS Vt-tile row stride (32 + 8 pad)
#define SSTR   40      // per-wave prob staging row stride

// double-buffered tiles: 2*(32*520 + 512*40) + 4*16*40 elems = 153,600 B
#define SMEM_BYTES ((2*KT*PSTR + 2*DIM*VSTR + NWAVE*16*SSTR) * 2)

union Frag { v16bf v; uint4 q[2]; };

// ------------------------------ prep kernel --------------------------------
__global__ __launch_bounds__(128)
void sa_prep(const float* __restrict__ pf, const float* __restrict__ qf,
             __bf16* __restrict__ qn, __bf16* __restrict__ pn,
             __bf16* __restrict__ vt)
{
    const int row   = blockIdx.x;          // 0 .. B*2048-1
    const int which = blockIdx.y;          // 0 = query, 1 = prompt
    const int b = row / NP;
    const int n = row % NP;
    const int t = threadIdx.x;             // 128 threads * 4 elems = 512

    const float* src = (which == 0 ? qf : pf) + (size_t)row * DIM;
    float4 x = ((const float4*)src)[t];
    float ss = x.x*x.x + x.y*x.y + x.z*x.z + x.w*x.w;
#pragma unroll
    for (int off = 16; off; off >>= 1) ss += __shfl_xor(ss, off);

    __shared__ float wsum[4];
    if ((t & 31) == 0) wsum[t >> 5] = ss;
    __syncthreads();
    float tot = wsum[0] + wsum[1] + wsum[2] + wsum[3];
    float inv = 1.0f / fmaxf(sqrtf(tot), 1e-12f);

    __bf16 e0 = (__bf16)(x.x * inv), e1 = (__bf16)(x.y * inv);
    __bf16 e2 = (__bf16)(x.z * inv), e3 = (__bf16)(x.w * inv);

    if (which == 0) {
        __bf16* dst = qn + (size_t)row * DIM + t * 4;
        dst[0] = e0; dst[1] = e1; dst[2] = e2; dst[3] = e3;
    } else {
        __bf16* dst = pn + (size_t)row * DIM + t * 4;
        dst[0] = e0; dst[1] = e1; dst[2] = e2; dst[3] = e3;
        // raw V transposed: vt[b][d][n]
        const int d0 = t * 4;
        __bf16* vd = vt + ((size_t)b * DIM) * NP + n;
        vd[(size_t)(d0+0)*NP] = (__bf16)x.x;
        vd[(size_t)(d0+1)*NP] = (__bf16)x.y;
        vd[(size_t)(d0+2)*NP] = (__bf16)x.z;
        vd[(size_t)(d0+3)*NP] = (__bf16)x.w;
    }
}

// --------------------------- flash attention kernel ------------------------
__global__ __launch_bounds__(128)
void sa_attn(const __bf16* __restrict__ qn, const __bf16* __restrict__ pn,
             const __bf16* __restrict__ vt, float* __restrict__ out)
{
    extern __shared__ __bf16 smem[];
    __bf16* lds_p  = smem;                              // 2 x [KT][PSTR]
    __bf16* lds_vt = smem + 2 * KT * PSTR;              // 2 x [DIM][VSTR]
    __bf16* lds_st = lds_vt + 2 * DIM * VSTR;           // [NWAVE][16][SSTR]

    const int b    = blockIdx.y;
    const int q0   = blockIdx.x * QT;
    const int tid  = threadIdx.x;
    const int wid  = tid >> 5;
    const int lane = tid & 31;
    const int ln   = lane & 15;
    const int g    = lane >> 4;

    // -------- preload Q A-fragments (16 K-steps of 32), loop-invariant ------
    // A layout: lane holds row m=ln; contiguous K run of 16 at 16g
    Frag qfrag[16];
    {
        const __bf16* qrow = qn + ((size_t)b * NQ + q0 + wid * 16 + ln) * DIM;
#pragma unroll
        for (int kk = 0; kk < 16; ++kk) {
            qfrag[kk].q[0] = *(const uint4*)(qrow + kk * 32 + g * 16);
            qfrag[kk].q[1] = *(const uint4*)(qrow + kk * 32 + g * 16 + 8);
        }
    }

    // async DMA of one prompt tile pair (P-norm KTxDIM, Vt DIMxKT) into buf
    // 32 x GLOBAL_LOAD_ASYNC_TO_LDS_B128 per thread (ASYNCcnt-tracked)
    auto issueTiles = [&](int j, int buf) {
        const __bf16* gp = pn + ((size_t)b * NP + j * KT) * DIM;
        unsigned lbP = (unsigned)(uintptr_t)(lds_p + buf * (KT * PSTR));
#pragma unroll
        for (int it = 0; it < 16; ++it) {
            int idx = it * 128 + tid;
            int r  = idx >> 6;
            int c8 = (idx & 63) * 8;
            unsigned laddr = lbP + (unsigned)(r * PSTR + c8) * 2u;
            unsigned voff  = (unsigned)(r * DIM + c8) * 2u;
            asm volatile("global_load_async_to_lds_b128 %0, %1, %2"
                         :: "v"(laddr), "v"(voff), "s"(gp) : "memory");
        }
        const __bf16* gv = vt + (size_t)b * DIM * NP + j * KT;
        unsigned lbV = (unsigned)(uintptr_t)(lds_vt + buf * (DIM * VSTR));
#pragma unroll
        for (int it = 0; it < 16; ++it) {
            int idx = it * 128 + tid;
            int r  = idx >> 2;
            int c8 = (idx & 3) * 8;
            unsigned laddr = lbV + (unsigned)(r * VSTR + c8) * 2u;
            unsigned voff  = (unsigned)(r * NP + c8) * 2u;   // < 2 MB, fits u32
            asm volatile("global_load_async_to_lds_b128 %0, %1, %2"
                         :: "v"(laddr), "v"(voff), "s"(gv) : "memory");
        }
    };

    v8f zero = {};
    v8f oacc[32];
#pragma unroll
    for (int i = 0; i < 32; ++i) oacc[i] = zero;
    float rowMax[8], rowSum[8];
#pragma unroll
    for (int v = 0; v < 8; ++v) { rowMax[v] = -__builtin_inff(); rowSum[v] = 0.f; }

    const float cinv = 1.0f / TAU;
    __bf16* stage = lds_st + wid * (16 * SSTR);
    const int NIT = NP / KT;   // 64

    issueTiles(0, 0);

    for (int j = 0; j < NIT; ++j) {
        const int cur = j & 1;
        __syncthreads();   // all waves done computing j-1 -> buf (cur^1) reusable
        if (j + 1 < NIT) {
            issueTiles(j + 1, cur ^ 1);
            // in-order async retirement: drain current batch, keep 32 in flight
            asm volatile("s_wait_asynccnt 0x20" ::: "memory");
        } else {
            asm volatile("s_wait_asynccnt 0x0" ::: "memory");
        }
        __syncthreads();   // every wave's current-buffer DMA has landed

        const __bf16* tp = lds_p  + cur * (KT * PSTR);
        const __bf16* tv = lds_vt + cur * (DIM * VSTR);

        // ---- S = Q (16x512) . P^T (512x32): two N-tiles, 16 K-steps -------
        // B layout: lane holds col n=ln (+16*t), contiguous K run of 16 at 16g
        v8f sac0 = zero, sac1 = zero;
#pragma unroll
        for (int kk = 0; kk < 16; ++kk) {
            Frag b0, b1;
            const __bf16* p0 = tp + (ln)      * PSTR + kk * 32 + g * 16;
            const __bf16* p1 = tp + (16 + ln) * PSTR + kk * 32 + g * 16;
            b0.q[0] = *(const uint4*)(p0); b0.q[1] = *(const uint4*)(p0 + 8);
            b1.q[0] = *(const uint4*)(p1); b1.q[1] = *(const uint4*)(p1 + 8);
            sac0 = __builtin_amdgcn_wmma_f32_16x16x32_bf16(
                false, qfrag[kk].v, false, b0.v, (short)0, sac0, false, false);
            sac1 = __builtin_amdgcn_wmma_f32_16x16x32_bf16(
                false, qfrag[kk].v, false, b1.v, (short)0, sac1, false, false);
        }

        // ---- online softmax; C layout: elem v of this lane is row v+8g ----
        float e0[8], e1[8], scale[8];
#pragma unroll
        for (int v = 0; v < 8; ++v) {
            float mx = fmaxf(sac0[v], sac1[v]);
#pragma unroll
            for (int off = 8; off; off >>= 1) mx = fmaxf(mx, __shfl_xor(mx, off));
            float nm = fmaxf(rowMax[v], mx);
            scale[v] = __expf((rowMax[v] - nm) * cinv);
            rowMax[v] = nm;
            e0[v] = __expf((sac0[v] - nm) * cinv);
            e1[v] = __expf((sac1[v] - nm) * cinv);
            float sm = e0[v] + e1[v];
#pragma unroll
            for (int off = 8; off; off >>= 1) sm += __shfl_xor(sm, off);
            rowSum[v] = rowSum[v] * scale[v] + sm;
        }
        // stage probabilities (row-major 16x32) so they re-enter in A layout
#pragma unroll
        for (int v = 0; v < 8; ++v) {
            int rm = v + 8 * g;
            stage[rm * SSTR + ln]      = (__bf16)e0[v];
            stage[rm * SSTR + ln + 16] = (__bf16)e1[v];
        }
        // wave-local LDS RAW across lanes: drain DS counter before reload
        asm volatile("s_wait_dscnt 0x0" ::: "memory");

        // rescale running O by exp(oldMax - newMax)
#pragma unroll
        for (int nt = 0; nt < 32; ++nt)
#pragma unroll
            for (int v = 0; v < 8; ++v) oacc[nt][v] *= scale[v];

        // ---- O += Ptilde (16x32) . V (32x512): 32 N-tiles, K=32 -----------
        Frag af;
        af.q[0] = *(const uint4*)(stage + ln * SSTR + g * 16);
        af.q[1] = *(const uint4*)(stage + ln * SSTR + g * 16 + 8);
#pragma unroll
        for (int nt = 0; nt < 32; ++nt) {
            Frag bv;
            const __bf16* vp = tv + (nt * 16 + ln) * VSTR + g * 16;
            bv.q[0] = *(const uint4*)(vp); bv.q[1] = *(const uint4*)(vp + 8);
            oacc[nt] = __builtin_amdgcn_wmma_f32_16x16x32_bf16(
                false, af.v, false, bv.v, (short)0, oacc[nt], false, false);
        }
    }

    // ---- finalize: divide by softmax denominator, store fp32 --------------
    float inv[8];
#pragma unroll
    for (int v = 0; v < 8; ++v) inv[v] = 1.0f / rowSum[v];
    const int m0 = q0 + wid * 16;
#pragma unroll
    for (int nt = 0; nt < 32; ++nt)
#pragma unroll
        for (int v = 0; v < 8; ++v) {
            int mr = m0 + v + 8 * g;
            int d  = nt * 16 + ln;
            out[((size_t)b * NQ + mr) * DIM + d] = oacc[nt][v] * inv[v];
        }
}

// ------------------------------- launcher ----------------------------------
extern "C" void kernel_launch(void* const* d_in, const int* in_sizes, int n_in,
                              void* d_out, int out_size, void* d_ws, size_t ws_size,
                              hipStream_t stream)
{
    const float* pf = (const float*)d_in[0];   // prompt_feats [8,2048,512]
    const float* qf = (const float*)d_in[1];   // query_feats  [8,2048,512]
    float* out = (float*)d_out;

    __bf16* qn = (__bf16*)d_ws;                         // 16.78 MB
    __bf16* pn = qn + (size_t)BDIM * NQ * DIM;          // 16.78 MB
    __bf16* vt = pn + (size_t)BDIM * NP * DIM;          // 16.78 MB (total ~50.3 MB)

    dim3 pgrid(BDIM * NP, 2);
    sa_prep<<<pgrid, 128, 0, stream>>>(pf, qf, qn, pn, vt);

    (void)hipFuncSetAttribute((const void*)sa_attn,
                              hipFuncAttributeMaxDynamicSharedMemorySize,
                              SMEM_BYTES);
    dim3 agrid(NQ / QT, BDIM);   // 32 x 8 = 256 workgroups
    sa_attn<<<agrid, 128, SMEM_BYTES, stream>>>(qn, pn, vt, out);
}
